// PointNet2TreeSegmentor_22084721836486
// MI455X (gfx1250) — matrix operand order repaired and checked by reading, not compiled, not run on hardware
//
#include <hip/hip_runtime.h>
#include <hip/hip_bf16.h>

typedef __attribute__((ext_vector_type(16))) _Float16 v16h;
typedef __attribute__((ext_vector_type(8)))  _Float16 v8h;
typedef __attribute__((ext_vector_type(8)))  float    v8f;

#define BN_EPS 1e-5f
#define KNBR   32

// ============================================================================
// WMMA fragment loaders (CDNA5 16x16x32 f16, wave32)
// A 16x32 f16: lane L(0..15)=row, halves 0..15 -> K=kh*16+0..15  (kh=lane>>4)
// B 32x16 f16 pre-packed so each lane does one contiguous 32B v16h load.
// C/D 16x16 f32: lane col = lane&15, vgpr r -> row r + 8*(lane>>4)
// ============================================================================
__device__ __forceinline__ v16h load_a_frag(const _Float16* A, int ld, int m0, int k0, int lane) {
  int m  = m0 + (lane & 15);
  int kh = lane >> 4;
  const _Float16* p = A + (size_t)m * ld + k0 + kh * 16;
  v8h lo = *(const v8h*)p;
  v8h hi = *(const v8h*)(p + 8);
  v16h r;
#pragma unroll
  for (int i = 0; i < 8; ++i) { r[i] = lo[i]; r[i + 8] = hi[i]; }
  return r;
}

__device__ __forceinline__ v16h load_b_frag(const _Float16* Wp, int Co, int kb, int n0, int lane) {
  int n  = n0 + (lane & 15);
  int kh = lane >> 4;
  const _Float16* p = Wp + ((size_t)kb * Co + n) * 32 + kh * 16;
  return *(const v16h*)p;  // 32-byte aligned
}

// ============================================================================
// Weight packer: f32 W[Ci,Co] -> f16 packed [CiP/32][Co][32], zero-padded K.
// Entry e = kh*16 + j  maps to K = kb*32 + kh*16 + j (B-matrix lane layout).
// ============================================================================
__global__ void k_pack_weight(const float* __restrict__ W, _Float16* __restrict__ Wp,
                              int Ci, int CiP, int Co) {
  int t = blockIdx.x * blockDim.x + threadIdx.x;
  int total = (CiP >> 5) * Co;
  if (t >= total) return;
  int kb = t / Co, n = t % Co;
  _Float16* dst = Wp + ((size_t)kb * Co + n) * 32;
#pragma unroll
  for (int e = 0; e < 32; ++e) {
    int kh = e >> 4, j = e & 15;
    int K  = kb * 32 + kh * 16 + j;
    dst[e] = (K < Ci) ? (_Float16)W[(size_t)K * Co + n] : (_Float16)0.0f;
  }
}

// Fold BatchNorm(eval)+bias: s = g*rsqrt(v+eps), t = (b-m)*s + be. Plain layer: s=1, t=b(or 0)
__global__ void k_scale_bias(const float* g, const float* be, const float* m, const float* v,
                             const float* b, float* s, float* t, int C) {
  int c = blockIdx.x * blockDim.x + threadIdx.x;
  if (c >= C) return;
  float bb = b ? b[c] : 0.0f;
  if (g) {
    float sc = g[c] * rsqrtf(v[c] + BN_EPS);
    s[c] = sc;
    t[c] = (bb - m[c]) * sc + be[c];
  } else {
    s[c] = 1.0f;
    t[c] = bb;
  }
}

// ============================================================================
// Block-tiled GEMM: out[R,Co] = act( (A_f16[R,CiP] x Wp) * s + t ).
// Block = 256 threads = 8 waves -> 128 rows x 64 cols.
// Per K-step the block's shared 32x64 B-tile (one contiguous 4 KB chunk of the
// packed weights) is staged into LDS with GLOBAL_LOAD_ASYNC_TO_LDS_B128,
// double-buffered via ASYNCcnt. Each wave reuses its A fragment for 4 WMMAs,
// and all 4 B fragments are fetched before the WMMA group so the matrix ops
// issue back-to-back behind a single DScnt wait.
// ============================================================================
__global__ void __launch_bounds__(256)
k_wmma_gemm(const _Float16* __restrict__ A, const _Float16* __restrict__ Wp,
            const float* __restrict__ s, const float* __restrict__ t,
            _Float16* __restrict__ outH, float* __restrict__ outF,
            int R, int CiP, int Co, int relu) {
  __shared__ _Float16 sB[2][32 * 64];   // 2 x 4 KB double buffer
  int tid  = threadIdx.x;
  int lane = tid & 31;
  int wv   = tid >> 5;                  // 0..7
  int nBn  = Co >> 6;
  int bm   = blockIdx.x / nBn, bn = blockIdx.x % nBn;
  int m0   = (bm << 7) + (wv << 4);
  int n0   = bn << 6;
  int nkb  = CiP >> 5;

  v8f acc[4] = {{}, {}, {}, {}};

  unsigned ldsBase = (unsigned)(size_t)&sB[0][0];
  auto stageB = [&](int kb, int buf) {
    // 256 threads x 16 B = 4 KB; one async-to-LDS instruction per wave.
    unsigned lds = ldsBase + (unsigned)(buf * (32 * 64 * 2)) + (unsigned)(tid * 16);
    unsigned long long ga =
        (unsigned long long)(size_t)(Wp + ((size_t)kb * Co + n0) * 32) +
        (unsigned long long)(tid * 16);
    asm volatile("global_load_async_to_lds_b128 %0, %1, off"
                 :: "v"(lds), "v"(ga) : "memory");
  };

  stageB(0, 0);
  for (int kb = 0; kb < nkb; ++kb) {
    int cur = kb & 1;
    if (kb + 1 < nkb) {
      stageB(kb + 1, cur ^ 1);
      asm volatile("s_wait_asynccnt 0x1" ::: "memory");  // oldest (cur) done
    } else {
      asm volatile("s_wait_asynccnt 0x0" ::: "memory");
    }
    __syncthreads();  // all waves' slices of sB[cur] visible
    v16h a = load_a_frag(A, CiP, m0, kb << 5, lane);
    v16h bf[4];
#pragma unroll
    for (int j = 0; j < 4; ++j) {
      const _Float16* p =
          &sB[cur][((j << 4) + (lane & 15)) * 32 + ((lane >> 4) << 4)];
      bf[j] = *(const v16h*)p;
    }
#pragma unroll
    for (int j = 0; j < 4; ++j)
      acc[j] = __builtin_amdgcn_wmma_f32_16x16x32_f16(false, a, false, bf[j],
                                                      (short)0, acc[j], false, false);
    __syncthreads();  // all reads of sB[cur] done before it is re-staged
  }

  int mBase = m0 + ((lane >> 4) << 3);
#pragma unroll
  for (int j = 0; j < 4; ++j) {
    int col = n0 + (j << 4) + (lane & 15);
    float sc = s ? s[col] : 1.0f;
    float sh = t ? t[col] : 0.0f;
#pragma unroll
    for (int r = 0; r < 8; ++r) {
      float v = acc[j][r] * sc + sh;
      if (relu) v = fmaxf(v, 0.0f);
      size_t o = (size_t)(mBase + r) * Co + col;
      if (outH) outH[o] = (_Float16)v; else outF[o] = v;
    }
  }
}

// ============================================================================
// GEMM + fused masked max-pool over KNBR=32 neighbor rows (SA final layer).
// 1 wave = one center x one 16-col tile; computes two 16x16 tiles (32 rows).
// ============================================================================
__global__ void k_wmma_gemm_pool(const _Float16* __restrict__ A, const _Float16* __restrict__ Wp,
                                 const float* __restrict__ bias, const int* __restrict__ nbr,
                                 float* __restrict__ out, int SB, int CiP, int Co) {
  int lane = threadIdx.x & 31;
  int wave = (blockIdx.x * blockDim.x + threadIdx.x) >> 5;
  int nT = Co >> 4;
  if (wave >= SB * nT) return;
  int cs = wave / nT, nt = wave % nT;
  int n0 = nt << 4;
  int rowBase = cs << 5;  // 32 rows per center
  v8f c0 = {}, c1 = {};
  int nkb = CiP >> 5;
  for (int kb = 0; kb < nkb; ++kb) {
    if (kb + 1 < nkb)
      __builtin_prefetch(Wp + ((size_t)(kb + 1) * Co + n0) * 32, 0, 0);
    v16h b  = load_b_frag(Wp, Co, kb, n0, lane);
    v16h a0 = load_a_frag(A, CiP, rowBase,      kb << 5, lane);
    v16h a1 = load_a_frag(A, CiP, rowBase + 16, kb << 5, lane);
    c0 = __builtin_amdgcn_wmma_f32_16x16x32_f16(false, a0, false, b, (short)0, c0, false, false);
    c1 = __builtin_amdgcn_wmma_f32_16x16x32_f16(false, a1, false, b, (short)0, c1, false, false);
  }
  int col = n0 + (lane & 15);
  float bv = bias ? bias[col] : 0.0f;
  const int* nb = nbr + (size_t)cs * KNBR;
  int rlo = (lane >> 4) << 3;
  float m = -3.0e38f;
#pragma unroll
  for (int r = 0; r < 8; ++r) {
    if (nb[rlo + r]      >= 0) m = fmaxf(m, c0[r] + bv);
    if (nb[16 + rlo + r] >= 0) m = fmaxf(m, c1[r] + bv);
  }
  m = fmaxf(m, __shfl_xor(m, 16, 32));   // combine row-halves (same column)
  if (lane < 16) out[(size_t)cs * Co + col] = m;
}

// ============================================================================
// Farthest point sampling: one block per cloud, sequential with LDS argmax.
// ============================================================================
#define FPS_T 256
__global__ void k_fps(const float* __restrict__ pos, int* __restrict__ idx,
                      float* __restrict__ dists, int N, int S) {
  int b = blockIdx.x;
  const float* P = pos + (size_t)b * N * 3;
  int*   I = idx   + (size_t)b * S;
  float* D = dists + (size_t)b * N;
  int tid = threadIdx.x;
  __shared__ float sv[FPS_T];
  __shared__ int   si[FPS_T];
  __shared__ float lp[3];
  for (int i = tid; i < N; i += FPS_T) D[i] = 3.4e38f;
  if (tid == 0) I[0] = 0;
  __syncthreads();
  int last = 0;
  for (int s = 1; s < S; ++s) {
    if (tid == 0) { lp[0] = P[last*3]; lp[1] = P[last*3+1]; lp[2] = P[last*3+2]; }
    __syncthreads();
    float lx = lp[0], ly = lp[1], lz = lp[2];
    float bestv = -1.0f; int besti = 0;
    for (int i = tid; i < N; i += FPS_T) {
      float dx = P[i*3] - lx, dy = P[i*3+1] - ly, dz = P[i*3+2] - lz;
      float d = dx*dx + dy*dy + dz*dz;
      float cur = D[i];
      if (d < cur) { cur = d; D[i] = d; }
      if (cur > bestv) { bestv = cur; besti = i; }
    }
    sv[tid] = bestv; si[tid] = besti;
    __syncthreads();
    for (int off = FPS_T / 2; off > 0; off >>= 1) {
      if (tid < off && sv[tid + off] > sv[tid]) { sv[tid] = sv[tid + off]; si[tid] = si[tid + off]; }
      __syncthreads();
    }
    last = si[0];
    if (tid == 0) I[s] = last;
    __syncthreads();
  }
}

__global__ void k_gather_centers(const float* __restrict__ pos, const int* __restrict__ idx,
                                 float* __restrict__ centers, int B, int S, int N) {
  int t = blockIdx.x * blockDim.x + threadIdx.x;
  if (t >= B * S) return;
  int b = t / S;
  int j = idx[t];
  const float* p = pos + ((size_t)b * N + j) * 3;
  centers[t*3] = p[0]; centers[t*3+1] = p[1]; centers[t*3+2] = p[2];
}

// K nearest within radius (thread per center); -1 marks invalid slots.
__global__ void k_radius_group(const float* __restrict__ pos, const float* __restrict__ centers,
                               int* __restrict__ nbr, int B, int S, int N, float r2) {
  int t = blockIdx.x * blockDim.x + threadIdx.x;
  if (t >= B * S) return;
  int b = t / S;
  const float* P = pos + (size_t)b * N * 3;
  float cx = centers[t*3], cy = centers[t*3+1], cz = centers[t*3+2];
  float bd[KNBR]; int bi[KNBR];
  int cnt = 0;
  for (int i = 0; i < N; ++i) {
    float dx = P[i*3] - cx, dy = P[i*3+1] - cy, dz = P[i*3+2] - cz;
    float d = dx*dx + dy*dy + dz*dz;
    if (d > r2) continue;
    if (cnt < KNBR) {
      int j = cnt++;
      while (j > 0 && bd[j-1] > d) { bd[j] = bd[j-1]; bi[j] = bi[j-1]; --j; }
      bd[j] = d; bi[j] = i;
    } else if (d < bd[KNBR-1]) {
      int j = KNBR - 1;
      while (j > 0 && bd[j-1] > d) { bd[j] = bd[j-1]; bi[j] = bi[j-1]; --j; }
      bd[j] = d; bi[j] = i;
    }
  }
  int* O = nbr + (size_t)t * KNBR;
  for (int k = 0; k < KNBR; ++k) O[k] = (k < cnt) ? bi[k] : -1;
}

// Build f16 SA-MLP input rows: [x_j (F), pos_j - center (3), zero-pad to CiP]
__global__ void k_build_sa_input(const float* __restrict__ x, const float* __restrict__ pos,
                                 const float* __restrict__ centers, const int* __restrict__ nbr,
                                 _Float16* __restrict__ A, int B, int S, int Nsrc, int F, int CiP) {
  int row = blockIdx.x * blockDim.x + threadIdx.x;
  int total = B * S * KNBR;
  if (row >= total) return;
  int cs = row >> 5;          // b*S + s
  int b = cs / S;
  int j = nbr[row];
  _Float16* dst = A + (size_t)row * CiP;
  if (j < 0) { for (int c = 0; c < CiP; ++c) dst[c] = (_Float16)0.0f; return; }
  const float* xs = x   + ((size_t)b * Nsrc + j) * F;
  const float* ps = pos + ((size_t)b * Nsrc + j) * 3;
  const float* ct = centers + (size_t)cs * 3;
  for (int c = 0; c < F; ++c) dst[c] = (_Float16)xs[c];
  for (int c = 0; c < 3; ++c) dst[F + c] = (_Float16)(ps[c] - ct[c]);
  for (int c = F + 3; c < CiP; ++c) dst[c] = (_Float16)0.0f;
}

// 3-NN (thread per destination point): indices + raw inverse-squared weights.
__global__ void k_knn3(const float* __restrict__ pd, const float* __restrict__ ps,
                       int* __restrict__ idx3, float* __restrict__ w3, int B, int Nd, int Ns) {
  int t = blockIdx.x * blockDim.x + threadIdx.x;
  if (t >= B * Nd) return;
  int b = t / Nd;
  const float* S_ = ps + (size_t)b * Ns * 3;
  float x = pd[t*3], y = pd[t*3+1], z = pd[t*3+2];
  float d0 = 3.4e38f, d1 = 3.4e38f, d2 = 3.4e38f;
  int i0 = 0, i1 = 0, i2 = 0;
  for (int i = 0; i < Ns; ++i) {
    float dx = S_[i*3] - x, dy = S_[i*3+1] - y, dz = S_[i*3+2] - z;
    float d = dx*dx + dy*dy + dz*dz;
    if (d < d0)      { d2 = d1; i2 = i1; d1 = d0; i1 = i0; d0 = d; i0 = i; }
    else if (d < d1) { d2 = d1; i2 = i1; d1 = d; i1 = i; }
    else if (d < d2) { d2 = d; i2 = i; }
  }
  idx3[t*3] = i0; idx3[t*3+1] = i1; idx3[t*3+2] = i2;
  w3[t*3]   = 1.0f / fmaxf(d0, 1e-16f);
  w3[t*3+1] = 1.0f / fmaxf(d1, 1e-16f);
  w3[t*3+2] = 1.0f / fmaxf(d2, 1e-16f);
}

// upn0 input rows: [knn_interp(x1) (C1), x0 (C0), zeros..CiP]; one block per row.
__global__ void k_build_upn0_input(const float* __restrict__ x1, const int* __restrict__ idx3,
                                   const float* __restrict__ w3, const float* __restrict__ x0,
                                   _Float16* __restrict__ A,
                                   int Sd, int Ss, int C1, int C0, int CiP) {
  int row = blockIdx.x;           // b*Sd + s
  int b = row / Sd;
  int tid = threadIdx.x;
  int i0 = idx3[row*3], i1 = idx3[row*3+1], i2 = idx3[row*3+2];
  float w0 = w3[row*3], w1 = w3[row*3+1], w2 = w3[row*3+2];
  float wn = 1.0f / (w0 + w1 + w2);
  const float* f0 = x1 + ((size_t)b * Ss + i0) * C1;
  const float* f1 = x1 + ((size_t)b * Ss + i1) * C1;
  const float* f2 = x1 + ((size_t)b * Ss + i2) * C1;
  _Float16* dst = A + (size_t)row * CiP;
  for (int c = tid; c < C1; c += blockDim.x)
    dst[c] = (_Float16)((f0[c]*w0 + f1[c]*w1 + f2[c]*w2) * wn);
  const float* xx = x0 + (size_t)row * C0;
  for (int c = tid; c < C0; c += blockDim.x) dst[C1 + c] = (_Float16)xx[c];
  for (int c = C1 + C0 + tid; c < CiP; c += blockDim.x) dst[c] = (_Float16)0.0f;
}

// upn1 input rows: [knn_interp(x2) (C2), x (3), zeros..CiP]; one block per row.
__global__ void k_build_upn1_input(const float* __restrict__ x2, const int* __restrict__ idx3,
                                   const float* __restrict__ w3, const float* __restrict__ x,
                                   _Float16* __restrict__ A,
                                   int Nd, int Ss, int C2, int CiP) {
  int row = blockIdx.x;           // b*Nd + n
  int b = row / Nd;
  int tid = threadIdx.x;
  int i0 = idx3[row*3], i1 = idx3[row*3+1], i2 = idx3[row*3+2];
  float w0 = w3[row*3], w1 = w3[row*3+1], w2 = w3[row*3+2];
  float wn = 1.0f / (w0 + w1 + w2);
  const float* f0 = x2 + ((size_t)b * Ss + i0) * C2;
  const float* f1 = x2 + ((size_t)b * Ss + i1) * C2;
  const float* f2 = x2 + ((size_t)b * Ss + i2) * C2;
  _Float16* dst = A + (size_t)row * CiP;
  for (int c = tid; c < C2; c += blockDim.x)
    dst[c] = (_Float16)((f0[c]*w0 + f1[c]*w1 + f2[c]*w2) * wn);
  if (tid < 3) dst[C2 + tid] = (_Float16)x[(size_t)row * 3 + tid];
  for (int c = C2 + 3 + tid; c < CiP; c += blockDim.x) dst[c] = (_Float16)0.0f;
}

// Tiny regression head [64->32->16->1], weights staged in LDS, thread per row.
__global__ void k_reg_head(const float* __restrict__ x3,
                           const float* W0, const float* be0, const float* g0,
                           const float* m0, const float* v0,
                           const float* W1, const float* be1, const float* g1,
                           const float* m1, const float* v1,
                           const float* W2, float* __restrict__ out, int R) {
  __shared__ float sW0[64*32], sW1[32*16], sW2[16];
  __shared__ float s0[32], t0[32], s1[16], t1[16];
  int tid = threadIdx.x;
  for (int i = tid; i < 64*32; i += blockDim.x) sW0[i] = W0[i];
  for (int i = tid; i < 32*16; i += blockDim.x) sW1[i] = W1[i];
  if (tid < 16) sW2[tid] = W2[tid];
  if (tid < 32) { float sc = g0[tid] * rsqrtf(v0[tid] + BN_EPS); s0[tid] = sc; t0[tid] = -m0[tid]*sc + be0[tid]; }
  if (tid < 16) { float sc = g1[tid] * rsqrtf(v1[tid] + BN_EPS); s1[tid] = sc; t1[tid] = -m1[tid]*sc + be1[tid]; }
  __syncthreads();
  int r = blockIdx.x * blockDim.x + tid;
  if (r >= R) return;
  const float* xi = x3 + (size_t)r * 64;
  float xr[64];
  for (int k = 0; k < 64; ++k) xr[k] = xi[k];
  float h1[32];
  for (int c = 0; c < 32; ++c) {
    float a = 0.0f;
    for (int k = 0; k < 64; ++k) a += xr[k] * sW0[k*32 + c];
    h1[c] = fmaxf(a * s0[c] + t0[c], 0.0f);
  }
  float h2[16];
  for (int c = 0; c < 16; ++c) {
    float a = 0.0f;
    for (int k = 0; k < 32; ++k) a += h1[k] * sW1[k*16 + c];
    h2[c] = fmaxf(a * s1[c] + t1[c], 0.0f);
  }
  float o = 0.0f;
  for (int k = 0; k < 16; ++k) o += h2[k] * sW2[k];
  out[r] = o;
}

// ============================================================================
// Host
// ============================================================================
struct LayerDef { int wi, bi, bei, gi, mi, vi, Ci, Co; };

static inline unsigned cdiv(long a, long b) { return (unsigned)((a + b - 1) / b); }

extern "C" void kernel_launch(void* const* d_in, const int* in_sizes, int n_in,
                              void* d_out, int out_size, void* d_ws, size_t ws_size,
                              hipStream_t stream) {
  (void)in_sizes; (void)n_in; (void)out_size; (void)ws_size;
  constexpr int Bb = 2, Nn = 8192, Ff = 3, S0 = 4096, S1 = 1024;

  const float* x   = (const float*)d_in[0];
  const float* pos = (const float*)d_in[1];

  // params flattened jax-tree style (top-level dict order x,pos,params; nested sorted keys):
  // reg(l0:W,be,g,m,v; l1:W,be,g,m,v; l2:W) then sa0,sa1,upn0,upn1.
  static const LayerDef L[12] = {
    {13,14,15,16,17,18,    6,   64},  // sa0 l0
    {19,20,21,22,23,24,   64,  128},  // sa0 l1
    {25,26,-1,-1,-1,-1,  128,  256},  // sa0 l2 (plain, bias)
    {27,28,29,30,31,32,  259,  512},  // sa1 l0
    {33,34,35,36,37,38,  512, 1024},  // sa1 l1
    {39,40,-1,-1,-1,-1, 1024, 2048},  // sa1 l2 (plain, bias)
    {41,-1,42,43,44,45, 2307, 1024},  // upn0 l0
    {46,-1,47,48,49,50, 1024,  512},  // upn0 l1
    {51,-1,-1,-1,-1,-1,  512,  256},  // upn0 l2 (plain, no bias)
    {52,-1,53,54,55,56,  262,  256},  // upn1 l0
    {57,-1,58,59,60,61,  256,  128},  // upn1 l1
    {62,-1,-1,-1,-1,-1,  128,   64},  // upn1 l2 (plain, no bias)
  };

  char* base = (char*)d_ws;
  size_t off = 0;
  auto alloc = [&](size_t bytes) -> void* {
    void* p = base + off;
    off = (off + bytes + 255) & ~(size_t)255;
    return p;
  };

  // ---- pack all weights + fold BN into scale/shift ----
  _Float16* Wp[12]; float* Ls[12]; float* Lt[12]; int CiP[12];
  for (int i = 0; i < 12; ++i) {
    CiP[i] = (L[i].Ci + 31) & ~31;
    Wp[i] = (_Float16*)alloc((size_t)CiP[i] * L[i].Co * sizeof(_Float16));
    Ls[i] = (float*)alloc(L[i].Co * sizeof(float));
    Lt[i] = (float*)alloc(L[i].Co * sizeof(float));
    int entries = (CiP[i] >> 5) * L[i].Co;
    k_pack_weight<<<cdiv(entries, 256), 256, 0, stream>>>(
        (const float*)d_in[L[i].wi], Wp[i], L[i].Ci, CiP[i], L[i].Co);
    k_scale_bias<<<cdiv(L[i].Co, 256), 256, 0, stream>>>(
        L[i].gi  >= 0 ? (const float*)d_in[L[i].gi]  : nullptr,
        L[i].bei >= 0 ? (const float*)d_in[L[i].bei] : nullptr,
        L[i].mi  >= 0 ? (const float*)d_in[L[i].mi]  : nullptr,
        L[i].vi  >= 0 ? (const float*)d_in[L[i].vi]  : nullptr,
        L[i].bi  >= 0 ? (const float*)d_in[L[i].bi]  : nullptr,
        Ls[i], Lt[i], L[i].Co);
  }

  // ---- persistent buffers ----
  float* dists    = (float*)alloc((size_t)Bb * Nn * sizeof(float));
  int*   idx0     = (int*)  alloc((size_t)Bb * S0 * sizeof(int));
  float* centers0 = (float*)alloc((size_t)Bb * S0 * 3 * sizeof(float));
  int*   nbr0     = (int*)  alloc((size_t)Bb * S0 * KNBR * sizeof(int));
  int*   idx1     = (int*)  alloc((size_t)Bb * S1 * sizeof(int));
  float* centers1 = (float*)alloc((size_t)Bb * S1 * 3 * sizeof(float));
  int*   nbr1     = (int*)  alloc((size_t)Bb * S1 * KNBR * sizeof(int));
  float* x0buf    = (float*)alloc((size_t)Bb * S0 * 256 * sizeof(float));
  float* x1buf    = (float*)alloc((size_t)Bb * S1 * 2048 * sizeof(float));
  float* x2buf    = (float*)alloc((size_t)Bb * S0 * 256 * sizeof(float));
  float* x3buf    = (float*)alloc((size_t)Bb * Nn * 64 * sizeof(float));
  int*   idx3a    = (int*)  alloc((size_t)Bb * S0 * 3 * sizeof(int));
  float* w3a      = (float*)alloc((size_t)Bb * S0 * 3 * sizeof(float));
  int*   idx3b    = (int*)  alloc((size_t)Bb * Nn * 3 * sizeof(int));
  float* w3b      = (float*)alloc((size_t)Bb * Nn * 3 * sizeof(float));

  // ---- ping-pong f16 activation arenas ----
  _Float16* bufP = (_Float16*)alloc((size_t)65536 * 1024 * sizeof(_Float16));  // 134 MB
  _Float16* bufQ = (_Float16*)alloc((size_t)65536 * 512  * sizeof(_Float16));  // 67 MB

  auto gemm = [&](const _Float16* A, int li, _Float16* oh, float* of, int R, int relu) {
    int blocks = (R >> 7) * (L[li].Co >> 6);   // 128x64 block tiles, exact
    k_wmma_gemm<<<blocks, 256, 0, stream>>>(
        A, Wp[li], Ls[li], Lt[li], oh, of, R, CiP[li], L[li].Co, relu);
  };
  auto gemm_pool = [&](const _Float16* A, int li, const int* nbr, float* out, int SB) {
    long waves = (long)SB * (L[li].Co >> 4);
    k_wmma_gemm_pool<<<cdiv(waves, 8), 256, 0, stream>>>(
        A, Wp[li], Lt[li], nbr, out, SB, CiP[li], L[li].Co);
  };

  // ================== Stage sa0 ==================
  k_fps<<<Bb, FPS_T, 0, stream>>>(pos, idx0, dists, Nn, S0);
  k_gather_centers<<<cdiv(Bb * S0, 256), 256, 0, stream>>>(pos, idx0, centers0, Bb, S0, Nn);
  k_radius_group<<<cdiv(Bb * S0, 256), 256, 0, stream>>>(pos, centers0, nbr0, Bb, S0, Nn, 0.04f);
  k_build_sa_input<<<cdiv((long)Bb * S0 * KNBR, 256), 256, 0, stream>>>(
      x, pos, centers0, nbr0, bufP, Bb, S0, Nn, Ff, 32);
  gemm(bufP, 0, bufQ, nullptr, Bb * S0 * KNBR, 1);   //  32 ->  64
  gemm(bufQ, 1, bufP, nullptr, Bb * S0 * KNBR, 1);   //  64 -> 128
  gemm_pool(bufP, 2, nbr0, x0buf, Bb * S0);          // 128 -> 256, max over K

  // ================== Stage sa1 ==================
  k_fps<<<Bb, FPS_T, 0, stream>>>(centers0, idx1, dists, S0, S1);
  k_gather_centers<<<cdiv(Bb * S1, 256), 256, 0, stream>>>(centers0, idx1, centers1, Bb, S1, S0);
  k_radius_group<<<cdiv(Bb * S1, 256), 256, 0, stream>>>(centers0, centers1, nbr1, Bb, S1, S0, 0.16f);
  k_build_sa_input<<<cdiv((long)Bb * S1 * KNBR, 256), 256, 0, stream>>>(
      x0buf, centers0, centers1, nbr1, bufP, Bb, S1, S0, 256, 288);
  gemm(bufP, 3, bufQ, nullptr, Bb * S1 * KNBR, 1);   // 288  ->  512
  gemm(bufQ, 4, bufP, nullptr, Bb * S1 * KNBR, 1);   // 512  -> 1024
  gemm_pool(bufP, 5, nbr1, x1buf, Bb * S1);          // 1024 -> 2048, max over K

  // ================== upn0 (interp 1024 -> 4096 + MLP) ==================
  k_knn3<<<cdiv(Bb * S0, 256), 256, 0, stream>>>(centers0, centers1, idx3a, w3a, Bb, S0, S1);
  k_build_upn0_input<<<Bb * S0, 256, 0, stream>>>(x1buf, idx3a, w3a, x0buf, bufP,
                                                  S0, S1, 2048, 256, 2336);
  gemm(bufP, 6, bufQ, nullptr, Bb * S0, 1);          // 2336 -> 1024
  gemm(bufQ, 7, bufP, nullptr, Bb * S0, 1);          // 1024 ->  512
  gemm(bufP, 8, nullptr, x2buf, Bb * S0, 0);         //  512 ->  256 (f32)

  // ================== upn1 (interp 4096 -> 8192 + MLP) ==================
  k_knn3<<<cdiv(Bb * Nn, 256), 256, 0, stream>>>(pos, centers0, idx3b, w3b, Bb, Nn, S0);
  k_build_upn1_input<<<Bb * Nn, 256, 0, stream>>>(x2buf, idx3b, w3b, x, bufP, Nn, S0, 256, 288);
  gemm(bufP, 9,  bufQ, nullptr, Bb * Nn, 1);         // 288 -> 256
  gemm(bufQ, 10, bufP, nullptr, Bb * Nn, 1);         // 256 -> 128
  gemm(bufP, 11, nullptr, x3buf, Bb * Nn, 0);        // 128 ->  64 (f32)

  // ================== regression head ==================
  k_reg_head<<<cdiv(Bb * Nn, 256), 256, 0, stream>>>(
      x3buf,
      (const float*)d_in[2], (const float*)d_in[3], (const float*)d_in[4],
      (const float*)d_in[5], (const float*)d_in[6],
      (const float*)d_in[7], (const float*)d_in[8], (const float*)d_in[9],
      (const float*)d_in[10], (const float*)d_in[11],
      (const float*)d_in[12], (float*)d_out, Bb * Nn);
}
